// KVAE_miss_73315091742865
// MI455X (gfx1250) — matrix operand order repaired and checked by reading, not compiled, not run on hardware
//
#include <hip/hip_runtime.h>
#include <hip/hip_bf16.h>

// ---------------- types ----------------
typedef __bf16 bhalf;
typedef __attribute__((ext_vector_type(16))) __bf16 v16bf;
typedef __attribute__((ext_vector_type(8)))  float  v8f;

// ---------------- problem dims ----------------
#define T_DIM 512
#define BS    256
#define XD    128
#define MD    128
#define AD    8
#define ZD    4
#define UD    9
#define KMIX  3
#define HL    50
#define HIDN  128
#define NOISE_TRANS 0.08f
#define NOISE_EMIS  0.03f
#define INIT_COV    20.0f

// ---------------- workspace layout (bytes), ~29.1 MB total ----------------
#define OFF_A      ((size_t)0)          // (T,BS,8)  f32
#define OFF_ALPHA  ((size_t)4194304)    // (T,BS,3)  f32
#define OFF_MUP    ((size_t)5767168)    // (T,BS,4)  f32
#define OFF_SIGP   ((size_t)7864320)    // (T,BS,16) f32
#define OFF_MUF    ((size_t)16252928)   // (T,BS,4)  f32
#define OFF_SIGF   ((size_t)18350080)   // (T,BS,16) f32
#define OFF_MUS    ((size_t)26738688)   // (T,BS,4)  f32
#define OFF_W1BF   ((size_t)28835840)   // enc_W1 bf16 (128x256)
#define OFF_W2BF   ((size_t)28901376)   // enc_W2 bf16 (128x128)
#define OFF_WMBF   ((size_t)28934144)   // W_mean bf16 padded (16x128)
#define OFF_DW1BF  ((size_t)28938240)   // dec_W1 bf16 padded (128x32)
#define OFF_DW2BF  ((size_t)28946432)   // dec_W2 bf16 (128x128)
#define OFF_GWBF   ((size_t)28979200)   // gen_W  bf16 (128x128)
#define OFF_LSTMB  ((size_t)29011968)   // bih+bhh f32 (200)

// ---------------- WMMA helpers ----------------
__device__ __forceinline__ v8f wmma_bf16(v16bf a, v16bf b, v8f c) {
  // D = A(16x32) * B(32x16) + C, f32 accumulate
  return __builtin_amdgcn_wmma_f32_16x16x32_bf16(false, a, false, b, (short)0, c,
                                                 false, false);
}

template <int E0>
__device__ __forceinline__ void cvt8(v16bf& f, const float* p) {
#pragma unroll
  for (int i = 0; i < 8; ++i) f[E0 + i] = (bhalf)p[i];
}

template <int E0>
__device__ __forceinline__ void ld8(v16bf& f, const bhalf* p) {
#pragma unroll
  for (int i = 0; i < 8; ++i) f[E0 + i] = p[i];
}

__device__ __forceinline__ float sigm(float x) { return 1.f / (1.f + expf(-x)); }

// ============================================================
// 0) weight prep: f32 -> bf16 (+padding), fused LSTM bias
// ============================================================
__global__ void prep_kernel(const float* __restrict__ encW1, const float* __restrict__ encW2,
                            const float* __restrict__ Wmean, const float* __restrict__ decW1,
                            const float* __restrict__ decW2, const float* __restrict__ genW,
                            const float* __restrict__ bih,   const float* __restrict__ bhh,
                            char* __restrict__ ws) {
  int tid = blockIdx.x * blockDim.x + threadIdx.x;
  bhalf* w1  = (bhalf*)(ws + OFF_W1BF);
  bhalf* w2  = (bhalf*)(ws + OFF_W2BF);
  bhalf* wm  = (bhalf*)(ws + OFF_WMBF);
  bhalf* dw1 = (bhalf*)(ws + OFF_DW1BF);
  bhalf* dw2 = (bhalf*)(ws + OFF_DW2BF);
  bhalf* gw  = (bhalf*)(ws + OFF_GWBF);
  float* bsum = (float*)(ws + OFF_LSTMB);

  if (tid < HIDN * 256) w1[tid] = (bhalf)encW1[tid];
  if (tid < HIDN * HIDN) {
    w2[tid]  = (bhalf)encW2[tid];
    dw2[tid] = (bhalf)decW2[tid];
    gw[tid]  = (bhalf)genW[tid];
  }
  if (tid < 16 * HIDN) {  // W_mean (8x128) padded to 16 rows
    int r = tid >> 7, c = tid & 127;
    wm[tid] = (bhalf)(r < AD ? Wmean[r * HIDN + c] : 0.f);
  }
  if (tid < HIDN * 32) {  // dec_W1 (128x8) padded to K=32
    int r = tid >> 5, c = tid & 31;
    dw1[tid] = (bhalf)(c < AD ? decW1[r * AD + c] : 0.f);
  }
  if (tid < 4 * HL) bsum[tid] = bih[tid] + bhh[tid];
}

// ============================================================
// 1) encoder: a = tanh(tanh([x|m]W1^T+b1)W2^T+b2)W_mean^T + b_mean + eps
//    128 threads (4 waves), each wave owns a 16-row tile. WMMA bf16.
//    a written in (T,BS,8) layout for the sequential stages.
// ============================================================
__global__ void __launch_bounds__(128) encoder_kernel(
    const float* __restrict__ x, const float* __restrict__ m, const float* __restrict__ eps,
    const bhalf* __restrict__ W1, const float* __restrict__ b1,
    const bhalf* __restrict__ W2, const float* __restrict__ b2,
    const bhalf* __restrict__ Wm, const float* __restrict__ bm,
    float* __restrict__ a_out) {
  __shared__ bhalf sh1[4 * 16 * HIDN];
  __shared__ bhalf sh2[4 * 16 * HIDN];
  const int lane = threadIdx.x & 31;
  const int wave = threadIdx.x >> 5;
  const int hh   = lane >> 4;   // which half of the wave (K-group select)
  const int lrow = lane & 15;   // A-matrix row / B,N column within tile
  const int row0 = (blockIdx.x * 4 + wave) * 16;
  const int r  = row0 + lrow;           // global row of flattened (BS*T)
  const int bb = r >> 9, tt = r & (T_DIM - 1);
  const float* xr = x + ((size_t)bb * T_DIM + tt) * XD;
  const float* mr = m + ((size_t)bb * T_DIM + tt) * MD;

  const v8f vzero = {0.f, 0.f, 0.f, 0.f, 0.f, 0.f, 0.f, 0.f};
  v8f acc[8];

  // ---- layer 1: (16x256) @ W1^T(256x128) ----
#pragma unroll
  for (int n = 0; n < 8; ++n) acc[n] = vzero;
#pragma unroll
  for (int kc = 0; kc < 8; ++kc) {
    const int g0 = kc * 32 + hh * 16;   // contiguous 8-col group (never crosses 128)
    const int g1 = g0 + 8;
    const float* p0 = (g0 < XD) ? (xr + g0) : (mr + (g0 - XD));
    const float* p1 = (g1 < XD) ? (xr + g1) : (mr + (g1 - XD));
    if (kc < 7) __builtin_prefetch(p0 + 32, 0, 3);
    v16bf af;
    cvt8<0>(af, p0);
    cvt8<8>(af, p1);
#pragma unroll
    for (int n = 0; n < 8; ++n) {
      v16bf bf = *(const v16bf*)(W1 + (size_t)(n * 16 + lrow) * 256 + kc * 32 + hh * 16);
      acc[n] = wmma_bf16(af, bf, acc[n]);
    }
  }
#pragma unroll
  for (int n = 0; n < 8; ++n) {
    const float bias = b1[n * 16 + lrow];
#pragma unroll
    for (int i = 0; i < 8; ++i) {
      const int rr = i + hh * 8;  // D-frag row within tile
      sh1[(wave * 16 + rr) * HIDN + n * 16 + lrow] = (bhalf)tanhf(acc[n][i] + bias);
    }
  }
  __syncthreads();

  // ---- layer 2: (16x128) @ W2^T(128x128) ----
#pragma unroll
  for (int n = 0; n < 8; ++n) acc[n] = vzero;
#pragma unroll
  for (int kc = 0; kc < 4; ++kc) {
    const bhalf* base = sh1 + (wave * 16 + lrow) * HIDN + kc * 32 + hh * 16;
    v16bf af;
    ld8<0>(af, base);
    ld8<8>(af, base + 8);
#pragma unroll
    for (int n = 0; n < 8; ++n) {
      v16bf bf = *(const v16bf*)(W2 + (size_t)(n * 16 + lrow) * HIDN + kc * 32 + hh * 16);
      acc[n] = wmma_bf16(af, bf, acc[n]);
    }
  }
#pragma unroll
  for (int n = 0; n < 8; ++n) {
    const float bias = b2[n * 16 + lrow];
#pragma unroll
    for (int i = 0; i < 8; ++i) {
      const int rr = i + hh * 8;
      sh2[(wave * 16 + rr) * HIDN + n * 16 + lrow] = (bhalf)tanhf(acc[n][i] + bias);
    }
  }
  __syncthreads();

  // ---- layer 3: (16x128) @ W_mean^T(128x16, rows 8..15 zero-padded) ----
  v8f acc3 = vzero;
#pragma unroll
  for (int kc = 0; kc < 4; ++kc) {
    const bhalf* base = sh2 + (wave * 16 + lrow) * HIDN + kc * 32 + hh * 16;
    v16bf af;
    ld8<0>(af, base);
    ld8<8>(af, base + 8);
    v16bf bf = *(const v16bf*)(Wm + (size_t)lrow * HIDN + kc * 32 + hh * 16);
    acc3 = wmma_bf16(af, bf, acc3);
  }
  if (lrow < AD) {  // only cols 0..7 are real outputs
    const float bias = bm[lrow];
#pragma unroll
    for (int i = 0; i < 8; ++i) {
      const int rr = row0 + i + hh * 8;
      const int b2i = rr >> 9, t2i = rr & (T_DIM - 1);
      const float av = acc3[i] + bias + eps[((size_t)b2i * T_DIM + t2i) * AD + lrow];
      a_out[((size_t)t2i * BS + b2i) * AD + lrow] = av;
    }
  }
}

// ============================================================
// 2) LSTM over a_{t-1} + alpha softmax. One workgroup per batch elem,
//    thread j<200 owns gate unit j (weights in registers), h/c/g in LDS.
// ============================================================
__global__ void __launch_bounds__(256) lstm_alpha_kernel(
    const float* __restrict__ a_seq,   // (T,BS,8)
    const float* __restrict__ a_init,  // (1,8)
    const float* __restrict__ Wih, const float* __restrict__ Whh,
    const float* __restrict__ bsum,    // 200 = bih+bhh
    const float* __restrict__ aW, const float* __restrict__ ab,
    float* __restrict__ alpha)         // (T,BS,3)
{
  const int b = blockIdx.x;
  const int j = threadIdx.x;
  __shared__ float h[HL], c[HL], g[4 * HL], ain[AD], gal[KMIX];

  float wih[AD], whh[HL], bj = 0.f, aw[HL], abj = 0.f;
  if (j < 4 * HL) {
#pragma unroll
    for (int i = 0; i < AD; ++i) wih[i] = Wih[j * AD + i];
#pragma unroll
    for (int i = 0; i < HL; ++i) whh[i] = Whh[j * HL + i];
    bj = bsum[j];
  }
  if (j < KMIX) {
#pragma unroll
    for (int i = 0; i < HL; ++i) aw[i] = aW[j * HL + i];
    abj = ab[j];
  }
  if (j < HL) { h[j] = 0.f; c[j] = 0.f; }
  __syncthreads();

  for (int t = 0; t < T_DIM; ++t) {
    if (j < AD)
      ain[j] = (t == 0) ? a_init[j] : a_seq[((size_t)(t - 1) * BS + b) * AD + j];
    __syncthreads();
    if (j < 4 * HL) {
      float gv = bj;
#pragma unroll
      for (int i = 0; i < AD; ++i) gv += ain[i] * wih[i];
#pragma unroll
      for (int i = 0; i < HL; ++i) gv += h[i] * whh[i];
      g[j] = gv;
    }
    __syncthreads();
    if (j < HL) {  // gate order: i, f, g, o
      const float ig = sigm(g[j]);
      const float fg = sigm(g[HL + j]);
      const float gg = tanhf(g[2 * HL + j]);
      const float og = sigm(g[3 * HL + j]);
      const float cn = fg * c[j] + ig * gg;
      c[j] = cn;
      h[j] = og * tanhf(cn);
    }
    __syncthreads();
    if (j < KMIX) {
      float s = abj;
#pragma unroll
      for (int i = 0; i < HL; ++i) s += h[i] * aw[i];
      gal[j] = s;
    }
    __syncthreads();
    if (j == 0) {
      const float mx = fmaxf(gal[0], fmaxf(gal[1], gal[2]));
      const float e0 = expf(gal[0] - mx), e1 = expf(gal[1] - mx), e2 = expf(gal[2] - mx);
      const float inv = 1.f / (e0 + e1 + e2);
      float* o = alpha + ((size_t)t * BS + b) * KMIX;
      o[0] = e0 * inv; o[1] = e1 * inv; o[2] = e2 * inv;
    }
    __syncthreads();
  }
}

// ============================================================
// 3) Kalman forward scan. One thread per batch element; Joseph form;
//    8x8 innovation covariance solved with Cholesky (SPD).
// ============================================================
__global__ void __launch_bounds__(32) kf_forward_kernel(
    const float* __restrict__ alpha, const float* __restrict__ a_seq,
    const float* __restrict__ u_ext,
    const float* __restrict__ Ap, const float* __restrict__ Bp, const float* __restrict__ Cp,
    float* __restrict__ mu_pred, float* __restrict__ Sig_pred,
    float* __restrict__ mu_filt, float* __restrict__ Sig_filt) {
  __shared__ float sA[KMIX * ZD * ZD], sB[KMIX * ZD * UD], sC[KMIX * AD * ZD];
  for (int i = threadIdx.x; i < KMIX * ZD * ZD; i += 32) sA[i] = Ap[i];
  for (int i = threadIdx.x; i < KMIX * ZD * UD; i += 32) sB[i] = Bp[i];
  for (int i = threadIdx.x; i < KMIX * AD * ZD; i += 32) sC[i] = Cp[i];
  __syncthreads();

  const int b = blockIdx.x * 32 + threadIdx.x;
  float mu[ZD], Sig[ZD][ZD];

  for (int t = 0; t < T_DIM; ++t) {
    const float* al = alpha + ((size_t)t * BS + b) * KMIX;
    const float a0 = al[0], a1 = al[1], a2 = al[2];
    float Am[ZD][ZD], Bm[ZD][UD], Cm[AD][ZD];
#pragma unroll
    for (int i = 0; i < ZD; ++i)
#pragma unroll
      for (int jj = 0; jj < ZD; ++jj)
        Am[i][jj] = a0 * sA[i * ZD + jj] + a1 * sA[16 + i * ZD + jj] + a2 * sA[32 + i * ZD + jj];
#pragma unroll
    for (int i = 0; i < ZD; ++i)
#pragma unroll
      for (int jj = 0; jj < UD; ++jj)
        Bm[i][jj] = a0 * sB[i * UD + jj] + a1 * sB[36 + i * UD + jj] + a2 * sB[72 + i * UD + jj];
#pragma unroll
    for (int i = 0; i < AD; ++i)
#pragma unroll
      for (int jj = 0; jj < ZD; ++jj)
        Cm[i][jj] = a0 * sC[i * ZD + jj] + a1 * sC[32 + i * ZD + jj] + a2 * sC[64 + i * ZD + jj];

    float mup[ZD], Sp[ZD][ZD];
    if (t == 0) {
#pragma unroll
      for (int i = 0; i < ZD; ++i) {
        mup[i] = 0.f;
#pragma unroll
        for (int jj = 0; jj < ZD; ++jj) Sp[i][jj] = (i == jj) ? INIT_COV : 0.f;
      }
    } else {
      float u[UD];
#pragma unroll
      for (int jj = 0; jj < AD; ++jj) u[jj] = a_seq[((size_t)(t - 1) * BS + b) * AD + jj];
      u[8] = u_ext[(size_t)b * T_DIM + t];
      float AS[ZD][ZD];
#pragma unroll
      for (int i = 0; i < ZD; ++i) {
        float s = 0.f;
#pragma unroll
        for (int k = 0; k < ZD; ++k) s += Am[i][k] * mu[k];
#pragma unroll
        for (int k = 0; k < UD; ++k) s += Bm[i][k] * u[k];
        mup[i] = s;
#pragma unroll
        for (int jj = 0; jj < ZD; ++jj) {
          float t1 = 0.f;
#pragma unroll
          for (int k = 0; k < ZD; ++k) t1 += Am[i][k] * Sig[k][jj];
          AS[i][jj] = t1;
        }
      }
#pragma unroll
      for (int i = 0; i < ZD; ++i)
#pragma unroll
        for (int jj = 0; jj < ZD; ++jj) {
          float t1 = 0.f;
#pragma unroll
          for (int k = 0; k < ZD; ++k) t1 += AS[i][k] * Am[jj][k];
          Sp[i][jj] = t1 + ((i == jj) ? NOISE_TRANS : 0.f);
        }
    }
    {
      float* mp = mu_pred + ((size_t)t * BS + b) * ZD;
      float* sp = Sig_pred + ((size_t)t * BS + b) * ZD * ZD;
#pragma unroll
      for (int i = 0; i < ZD; ++i) mp[i] = mup[i];
#pragma unroll
      for (int i = 0; i < ZD; ++i)
#pragma unroll
        for (int jj = 0; jj < ZD; ++jj) sp[i * ZD + jj] = Sp[i][jj];
    }

    // ---- measurement update ----
    float res[AD];
#pragma unroll
    for (int i = 0; i < AD; ++i) {
      float s = a_seq[((size_t)t * BS + b) * AD + i];
#pragma unroll
      for (int jj = 0; jj < ZD; ++jj) s -= Cm[i][jj] * mup[jj];
      res[i] = s;
    }
    float CS[AD][ZD];
#pragma unroll
    for (int i = 0; i < AD; ++i)
#pragma unroll
      for (int jj = 0; jj < ZD; ++jj) {
        float s = 0.f;
#pragma unroll
        for (int k = 0; k < ZD; ++k) s += Cm[i][k] * Sp[k][jj];
        CS[i][jj] = s;
      }
    float S[AD][AD];
#pragma unroll
    for (int i = 0; i < AD; ++i)
#pragma unroll
      for (int jj = 0; jj < AD; ++jj) {
        if (jj <= i) {
          float s = 0.f;
#pragma unroll
          for (int k = 0; k < ZD; ++k) s += CS[i][k] * Cm[jj][k];
          S[i][jj] = s + ((i == jj) ? NOISE_EMIS : 0.f);
        } else {
          S[i][jj] = 0.f;
        }
      }
    // Cholesky (lower, in place)
#pragma unroll
    for (int k = 0; k < AD; ++k) {
      const float d = sqrtf(S[k][k]);
      S[k][k] = d;
      const float invd = 1.f / d;
      for (int i = k + 1; i < AD; ++i) S[i][k] *= invd;
      for (int jj = k + 1; jj < AD; ++jj)
        for (int i = jj; i < AD; ++i) S[i][jj] -= S[i][k] * S[jj][k];
    }
    // Kt = S^{-1} * CS   (so Kg = Kt^T)
    float Kt[AD][ZD];
#pragma unroll
    for (int col = 0; col < ZD; ++col) {
      float y[AD];
#pragma unroll
      for (int i = 0; i < AD; ++i) {
        float s = CS[i][col];
        for (int k = 0; k < i; ++k) s -= S[i][k] * y[k];
        y[i] = s / S[i][i];
      }
#pragma unroll
      for (int i = AD - 1; i >= 0; --i) {
        float s = y[i];
        for (int k = i + 1; k < AD; ++k) s -= S[k][i] * Kt[k][col];
        Kt[i][col] = s / S[i][i];
      }
    }
    float muf[ZD];
#pragma unroll
    for (int i = 0; i < ZD; ++i) {
      float s = mup[i];
#pragma unroll
      for (int jj = 0; jj < AD; ++jj) s += Kt[jj][i] * res[jj];
      muf[i] = s;
    }
    float IKC[ZD][ZD];
#pragma unroll
    for (int i = 0; i < ZD; ++i)
#pragma unroll
      for (int jj = 0; jj < ZD; ++jj) {
        float s = (i == jj) ? 1.f : 0.f;
#pragma unroll
        for (int k = 0; k < AD; ++k) s -= Kt[k][i] * Cm[k][jj];
        IKC[i][jj] = s;
      }
    float T1[ZD][ZD];
#pragma unroll
    for (int i = 0; i < ZD; ++i)
#pragma unroll
      for (int jj = 0; jj < ZD; ++jj) {
        float s = 0.f;
#pragma unroll
        for (int k = 0; k < ZD; ++k) s += IKC[i][k] * Sp[k][jj];
        T1[i][jj] = s;
      }
    float Sf[ZD][ZD];
#pragma unroll
    for (int i = 0; i < ZD; ++i)
#pragma unroll
      for (int jj = 0; jj < ZD; ++jj) {
        float s = 0.f, kk = 0.f;
#pragma unroll
        for (int k = 0; k < ZD; ++k) s += T1[i][k] * IKC[jj][k];
#pragma unroll
        for (int k = 0; k < AD; ++k) kk += Kt[k][i] * Kt[k][jj];
        Sf[i][jj] = s + NOISE_EMIS * kk;
      }
    {
      float* mf = mu_filt + ((size_t)t * BS + b) * ZD;
      float* sf = Sig_filt + ((size_t)t * BS + b) * ZD * ZD;
#pragma unroll
      for (int i = 0; i < ZD; ++i) { mf[i] = muf[i]; mu[i] = muf[i]; }
#pragma unroll
      for (int i = 0; i < ZD; ++i)
#pragma unroll
        for (int jj = 0; jj < ZD; ++jj) { sf[i * ZD + jj] = Sf[i][jj]; Sig[i][jj] = Sf[i][jj]; }
    }
  }
}

// ============================================================
// 4) RTS smoother (backward). 4x4 Cholesky solve per step per batch.
// ============================================================
__global__ void __launch_bounds__(32) rts_kernel(
    const float* __restrict__ alpha, const float* __restrict__ Ap,
    const float* __restrict__ mu_pred, const float* __restrict__ Sig_pred,
    const float* __restrict__ mu_filt, const float* __restrict__ Sig_filt,
    float* __restrict__ mu_s_out) {
  __shared__ float sA[KMIX * ZD * ZD];
  for (int i = threadIdx.x; i < KMIX * ZD * ZD; i += 32) sA[i] = Ap[i];
  __syncthreads();

  const int b = blockIdx.x * 32 + threadIdx.x;
  float mus[ZD], Sgs[ZD][ZD];
  {
    const float* mf = mu_filt + ((size_t)(T_DIM - 1) * BS + b) * ZD;
    const float* sf = Sig_filt + ((size_t)(T_DIM - 1) * BS + b) * ZD * ZD;
    float* mo = mu_s_out + ((size_t)(T_DIM - 1) * BS + b) * ZD;
#pragma unroll
    for (int i = 0; i < ZD; ++i) { mus[i] = mf[i]; mo[i] = mf[i]; }
#pragma unroll
    for (int i = 0; i < ZD; ++i)
#pragma unroll
      for (int jj = 0; jj < ZD; ++jj) Sgs[i][jj] = sf[i * ZD + jj];
  }

  for (int t = T_DIM - 2; t >= 0; --t) {
    const float* al = alpha + ((size_t)(t + 1) * BS + b) * KMIX;
    const float a0 = al[0], a1 = al[1], a2 = al[2];
    float Am[ZD][ZD];
#pragma unroll
    for (int i = 0; i < ZD; ++i)
#pragma unroll
      for (int jj = 0; jj < ZD; ++jj)
        Am[i][jj] = a0 * sA[i * ZD + jj] + a1 * sA[16 + i * ZD + jj] + a2 * sA[32 + i * ZD + jj];

    float Sf[ZD][ZD], Sp[ZD][ZD], muf[ZD], mup[ZD];
    {
      const float* p1 = Sig_filt + ((size_t)t * BS + b) * ZD * ZD;
      const float* p2 = Sig_pred + ((size_t)(t + 1) * BS + b) * ZD * ZD;
      const float* p3 = mu_filt + ((size_t)t * BS + b) * ZD;
      const float* p4 = mu_pred + ((size_t)(t + 1) * BS + b) * ZD;
#pragma unroll
      for (int i = 0; i < ZD; ++i) {
        muf[i] = p3[i]; mup[i] = p4[i];
#pragma unroll
        for (int jj = 0; jj < ZD; ++jj) { Sf[i][jj] = p1[i * ZD + jj]; Sp[i][jj] = p2[i * ZD + jj]; }
      }
    }
    // Mt = Am * Sf  (note J^T = Sp^{-1} * Am * Sf since Sf, Sp symmetric)
    float Mt[ZD][ZD];
#pragma unroll
    for (int i = 0; i < ZD; ++i)
#pragma unroll
      for (int jj = 0; jj < ZD; ++jj) {
        float s = 0.f;
#pragma unroll
        for (int k = 0; k < ZD; ++k) s += Am[i][k] * Sf[k][jj];
        Mt[i][jj] = s;
      }
    // Cholesky of Sp (lower in place)
#pragma unroll
    for (int k = 0; k < ZD; ++k) {
      const float d = sqrtf(Sp[k][k]);
      Sp[k][k] = d;
      const float invd = 1.f / d;
      for (int i = k + 1; i < ZD; ++i) Sp[i][k] *= invd;
      for (int jj = k + 1; jj < ZD; ++jj)
        for (int i = jj; i < ZD; ++i) Sp[i][jj] -= Sp[i][k] * Sp[jj][k];
    }
    float Jt[ZD][ZD];  // Jt = Sp^{-1} * Mt, so J[i][j] = Jt[j][i]
#pragma unroll
    for (int col = 0; col < ZD; ++col) {
      float y[ZD];
#pragma unroll
      for (int i = 0; i < ZD; ++i) {
        float s = Mt[i][col];
        for (int k = 0; k < i; ++k) s -= Sp[i][k] * y[k];
        y[i] = s / Sp[i][i];
      }
#pragma unroll
      for (int i = ZD - 1; i >= 0; --i) {
        float s = y[i];
        for (int k = i + 1; k < ZD; ++k) s -= Sp[k][i] * Jt[k][col];
        Jt[i][col] = s / Sp[i][i];
      }
    }
    float dmu[ZD], musn[ZD];
#pragma unroll
    for (int i = 0; i < ZD; ++i) dmu[i] = mus[i] - mup[i];
#pragma unroll
    for (int i = 0; i < ZD; ++i) {
      float s = muf[i];
#pragma unroll
      for (int jj = 0; jj < ZD; ++jj) s += Jt[jj][i] * dmu[jj];
      musn[i] = s;
    }
    float DD[ZD][ZD], T2[ZD][ZD], Sn[ZD][ZD];
#pragma unroll
    for (int i = 0; i < ZD; ++i)
#pragma unroll
      for (int jj = 0; jj < ZD; ++jj) DD[i][jj] = Sgs[i][jj] - ((i >= jj) ? 0.f : 0.f) - 0.f;
    // DD = Sgs - Sp_original; Sp was destroyed by Cholesky, reload:
    {
      const float* p2 = Sig_pred + ((size_t)(t + 1) * BS + b) * ZD * ZD;
#pragma unroll
      for (int i = 0; i < ZD; ++i)
#pragma unroll
        for (int jj = 0; jj < ZD; ++jj) DD[i][jj] = Sgs[i][jj] - p2[i * ZD + jj];
    }
#pragma unroll
    for (int k = 0; k < ZD; ++k)
#pragma unroll
      for (int jj = 0; jj < ZD; ++jj) {
        float s = 0.f;
#pragma unroll
        for (int l = 0; l < ZD; ++l) s += DD[k][l] * Jt[l][jj];
        T2[k][jj] = s;
      }
#pragma unroll
    for (int i = 0; i < ZD; ++i)
#pragma unroll
      for (int jj = 0; jj < ZD; ++jj) {
        float s = Sf[i][jj];
#pragma unroll
        for (int k = 0; k < ZD; ++k) s += Jt[k][i] * T2[k][jj];
        Sn[i][jj] = s;
      }
    float* mo = mu_s_out + ((size_t)t * BS + b) * ZD;
#pragma unroll
    for (int i = 0; i < ZD; ++i) { mus[i] = musn[i]; mo[i] = musn[i]; }
#pragma unroll
    for (int i = 0; i < ZD; ++i)
#pragma unroll
      for (int jj = 0; jj < ZD; ++jj) Sgs[i][jj] = Sn[i][jj];
  }
}

// ============================================================
// 5) decoder: a_hat = C_mix mu_smooth -> tanh MLP -> sigmoid. WMMA bf16.
//    dec_W1 is zero-padded to K=32 so layer 1 also runs on WMMA.
// ============================================================
__global__ void __launch_bounds__(128) decoder_kernel(
    const float* __restrict__ alpha, const float* __restrict__ Cp,
    const float* __restrict__ mu_s,
    const bhalf* __restrict__ DW1, const float* __restrict__ db1,
    const bhalf* __restrict__ DW2, const float* __restrict__ db2,
    const bhalf* __restrict__ GW, const float* __restrict__ gb,
    float* __restrict__ out) {
  __shared__ bhalf sh1[4 * 16 * HIDN];
  __shared__ bhalf sh2[4 * 16 * HIDN];
  const int lane = threadIdx.x & 31;
  const int wave = threadIdx.x >> 5;
  const int hh   = lane >> 4;
  const int lrow = lane & 15;
  const int row0 = (blockIdx.x * 4 + wave) * 16;
  const int r  = row0 + lrow;  // rows ordered (b, t) to match output layout
  const int bb = r >> 9, tt = r & (T_DIM - 1);

  const v8f vzero = {0.f, 0.f, 0.f, 0.f, 0.f, 0.f, 0.f, 0.f};

  // a_hat (8 values, padded to K=32). Only half==0 lanes carry data.
  v16bf af1;
#pragma unroll
  for (int i = 0; i < 16; ++i) af1[i] = (bhalf)0.f;
  if (hh == 0) {
    const float* al = alpha + ((size_t)tt * BS + bb) * KMIX;
    const float a0 = al[0], a1 = al[1], a2 = al[2];
    const float* mv = mu_s + ((size_t)tt * BS + bb) * ZD;
    float muv[ZD];
#pragma unroll
    for (int jj = 0; jj < ZD; ++jj) muv[jj] = mv[jj];
#pragma unroll
    for (int i = 0; i < AD; ++i) {
      float s = 0.f;
#pragma unroll
      for (int jj = 0; jj < ZD; ++jj) {
        const float cmix = a0 * Cp[i * ZD + jj] + a1 * Cp[32 + i * ZD + jj] + a2 * Cp[64 + i * ZD + jj];
        s += cmix * muv[jj];
      }
      af1[i] = (bhalf)s;
    }
  }

  v8f acc[8];
  // ---- layer 1: (16x32 padded) @ dec_W1^T ----
#pragma unroll
  for (int n = 0; n < 8; ++n) {
    v16bf bf = *(const v16bf*)(DW1 + (size_t)(n * 16 + lrow) * 32 + hh * 16);
    acc[n] = wmma_bf16(af1, bf, vzero);
  }
#pragma unroll
  for (int n = 0; n < 8; ++n) {
    const float bias = db1[n * 16 + lrow];
#pragma unroll
    for (int i = 0; i < 8; ++i) {
      const int rr = i + hh * 8;
      sh1[(wave * 16 + rr) * HIDN + n * 16 + lrow] = (bhalf)tanhf(acc[n][i] + bias);
    }
  }
  __syncthreads();

  // ---- layer 2: (16x128) @ dec_W2^T ----
#pragma unroll
  for (int n = 0; n < 8; ++n) acc[n] = vzero;
#pragma unroll
  for (int kc = 0; kc < 4; ++kc) {
    const bhalf* base = sh1 + (wave * 16 + lrow) * HIDN + kc * 32 + hh * 16;
    v16bf af;
    ld8<0>(af, base);
    ld8<8>(af, base + 8);
#pragma unroll
    for (int n = 0; n < 8; ++n) {
      v16bf bf = *(const v16bf*)(DW2 + (size_t)(n * 16 + lrow) * HIDN + kc * 32 + hh * 16);
      acc[n] = wmma_bf16(af, bf, acc[n]);
    }
  }
#pragma unroll
  for (int n = 0; n < 8; ++n) {
    const float bias = db2[n * 16 + lrow];
#pragma unroll
    for (int i = 0; i < 8; ++i) {
      const int rr = i + hh * 8;
      sh2[(wave * 16 + rr) * HIDN + n * 16 + lrow] = (bhalf)tanhf(acc[n][i] + bias);
    }
  }
  __syncthreads();

  // ---- layer 3: (16x128) @ gen_W^T, sigmoid, write m_mean (bs,T,128) ----
#pragma unroll
  for (int n = 0; n < 8; ++n) acc[n] = vzero;
#pragma unroll
  for (int kc = 0; kc < 4; ++kc) {
    const bhalf* base = sh2 + (wave * 16 + lrow) * HIDN + kc * 32 + hh * 16;
    v16bf af;
    ld8<0>(af, base);
    ld8<8>(af, base + 8);
#pragma unroll
    for (int n = 0; n < 8; ++n) {
      v16bf bf = *(const v16bf*)(GW + (size_t)(n * 16 + lrow) * HIDN + kc * 32 + hh * 16);
      acc[n] = wmma_bf16(af, bf, acc[n]);
    }
  }
#pragma unroll
  for (int n = 0; n < 8; ++n) {
    const float bias = gb[n * 16 + lrow];
#pragma unroll
    for (int i = 0; i < 8; ++i) {
      const int rr = row0 + i + hh * 8;
      const int b2i = rr >> 9, t2i = rr & (T_DIM - 1);
      out[((size_t)b2i * T_DIM + t2i) * MD + n * 16 + lrow] = sigm(acc[n][i] + bias);
    }
  }
}

// ============================================================
// host launcher
// ============================================================
extern "C" void kernel_launch(void* const* d_in, const int* in_sizes, int n_in,
                              void* d_out, int out_size, void* d_ws, size_t ws_size,
                              hipStream_t stream) {
  (void)in_sizes; (void)n_in; (void)out_size; (void)ws_size;
  const float* x      = (const float*)d_in[0];
  const float* m      = (const float*)d_in[1];
  const float* u_ext  = (const float*)d_in[2];
  const float* eps    = (const float*)d_in[3];
  const float* enc_W1 = (const float*)d_in[4];
  const float* enc_b1 = (const float*)d_in[5];
  const float* enc_W2 = (const float*)d_in[6];
  const float* enc_b2 = (const float*)d_in[7];
  const float* W_mean = (const float*)d_in[8];
  const float* b_mean = (const float*)d_in[9];
  const float* Ap     = (const float*)d_in[10];
  const float* Bp     = (const float*)d_in[11];
  const float* Cp     = (const float*)d_in[12];
  const float* a_init = (const float*)d_in[13];
  const float* Wih    = (const float*)d_in[14];
  const float* Whh    = (const float*)d_in[15];
  const float* bih    = (const float*)d_in[16];
  const float* bhh    = (const float*)d_in[17];
  const float* aW     = (const float*)d_in[18];
  const float* ab     = (const float*)d_in[19];
  const float* dW1    = (const float*)d_in[20];
  const float* db1    = (const float*)d_in[21];
  const float* dW2    = (const float*)d_in[22];
  const float* db2    = (const float*)d_in[23];
  const float* gW     = (const float*)d_in[24];
  const float* gb     = (const float*)d_in[25];
  char* ws = (char*)d_ws;
  float* out = (float*)d_out;

  prep_kernel<<<128, 256, 0, stream>>>(enc_W1, enc_W2, W_mean, dW1, dW2, gW, bih, bhh, ws);

  encoder_kernel<<<2048, 128, 0, stream>>>(
      x, m, eps,
      (const bhalf*)(ws + OFF_W1BF), enc_b1,
      (const bhalf*)(ws + OFF_W2BF), enc_b2,
      (const bhalf*)(ws + OFF_WMBF), b_mean,
      (float*)(ws + OFF_A));

  lstm_alpha_kernel<<<256, 256, 0, stream>>>(
      (const float*)(ws + OFF_A), a_init, Wih, Whh,
      (const float*)(ws + OFF_LSTMB), aW, ab,
      (float*)(ws + OFF_ALPHA));

  kf_forward_kernel<<<8, 32, 0, stream>>>(
      (const float*)(ws + OFF_ALPHA), (const float*)(ws + OFF_A), u_ext,
      Ap, Bp, Cp,
      (float*)(ws + OFF_MUP), (float*)(ws + OFF_SIGP),
      (float*)(ws + OFF_MUF), (float*)(ws + OFF_SIGF));

  rts_kernel<<<8, 32, 0, stream>>>(
      (const float*)(ws + OFF_ALPHA), Ap,
      (const float*)(ws + OFF_MUP), (const float*)(ws + OFF_SIGP),
      (const float*)(ws + OFF_MUF), (const float*)(ws + OFF_SIGF),
      (float*)(ws + OFF_MUS));

  decoder_kernel<<<2048, 128, 0, stream>>>(
      (const float*)(ws + OFF_ALPHA), Cp,
      (const float*)(ws + OFF_MUS),
      (const bhalf*)(ws + OFF_DW1BF), db1,
      (const bhalf*)(ws + OFF_DW2BF), db2,
      (const bhalf*)(ws + OFF_GWBF), gb,
      out);
}
